// PyTorchProjectionPlusCrossEntropyLoss_68315749810211
// MI455X (gfx1250) — matrix operand order repaired and checked by reading, not compiled, not run on hardware
//
#include <hip/hip_runtime.h>
#include <hip/hip_bf16.h>

// ---------------------------------------------------------------------------
// Fused projection + cross-entropy (mean NLL, ignore_index=-100) for gfx1250.
//   logits = x[8192,2048] @ W[50257,2048]^T + b   ->  mean NLL at targ
// HBM-bound: stream W once (412MB ~ 18us @ 23.3TB/s). bf16 WMMA 16x16x32 with
// fp32 accumulation; operands staged in LDS (converted to bf16 once, double
// buffered) so global traffic per block is exactly the unique data; fused
// online softmax so logits never touch memory.
// ---------------------------------------------------------------------------

#define N_ROWS      8192
#define K_DIM       2048
#define V_SIZE      50257
#define IGNORE_IDX  (-100)

#define CHUNK       512                          // vocab cols per block
#define SUBW        128                          // vocab cols per sub-iter
#define NSUB        (CHUNK / SUBW)               // 4
#define NCHUNKS     ((V_SIZE + CHUNK - 1) / CHUNK)  // 99
#define BLOCK_ROWS  128
#define NROWBLKS    (N_ROWS / BLOCK_ROWS)        // 64
#define RED_BLOCKS  (N_ROWS / 256)               // 32

// LDS slab row stride in bf16 elements: 32 K values + 8 pad.
// 80B row pitch = 20 banks -> the 16-lane ds_load_b128 pattern covers all 64
// banks conflict-free (20*ln mod 64 enumerates {0,4,...,60}).
#define LDS_STRIDE  40

typedef __bf16 bf16_t;
typedef __attribute__((ext_vector_type(16))) __bf16 v16bf;
typedef __attribute__((ext_vector_type(8)))  float  v8f;

union Pack16 { bf16_t h[16]; uint4 q[2]; };
struct F16R  { float4 f[4]; };

__device__ __forceinline__ v8f wmma_bf16(v16bf a, v16bf b, v8f c) {
  // D = A(16x32) x B(32x16) + C, fp32 accum.
  return __builtin_amdgcn_wmma_f32_16x16x32_bf16(
      /*neg_a=*/false, a, /*neg_b=*/false, b,
      /*c_mod=*/(short)0, c, /*reuse_a=*/false, /*reuse_b=*/false);
}

__device__ __forceinline__ F16R load16(const float* __restrict__ p) {
  F16R r;
#pragma unroll
  for (int i = 0; i < 4; ++i) r.f[i] = *reinterpret_cast<const float4*>(p + 4 * i);
  return r;
}

__device__ __forceinline__ Pack16 pack16(const F16R& r) {
  Pack16 u;
#pragma unroll
  for (int i = 0; i < 4; ++i) {
    u.h[4 * i + 0] = (bf16_t)r.f[i].x; u.h[4 * i + 1] = (bf16_t)r.f[i].y;
    u.h[4 * i + 2] = (bf16_t)r.f[i].z; u.h[4 * i + 3] = (bf16_t)r.f[i].w;
  }
  return u;
}

__device__ __forceinline__ void store_pack(bf16_t* d, const Pack16& u) {
  uint4* q = reinterpret_cast<uint4*>(d);
  q[0] = u.q[0];
  q[1] = u.q[1];
}

// Two 16B LDS reads -> one v16bf WMMA operand (offsets in bf16 elements).
__device__ __forceinline__ v16bf lds_read(const bf16_t* p0, const bf16_t* p1) {
  union { uint4 q[2]; v16bf v; } u;
  u.q[0] = *reinterpret_cast<const uint4*>(p0);
  u.q[1] = *reinterpret_cast<const uint4*>(p1);
  return u.v;
}

__device__ __forceinline__ v8f vzero8() {
  v8f z;
#pragma unroll
  for (int i = 0; i < 8; ++i) z[i] = 0.f;
  return z;
}

// ---------------------------------------------------------------------------
// Kernel 1: per-(rowblock, vocab-chunk) GEMM tile + online softmax partials.
// Block = 256 threads = 8 waves laid out 4(M) x 2(N); wave tile = 32M x 64N.
// Operands double-buffered in LDS as bf16 (converted once at fill time).
// ---------------------------------------------------------------------------
__global__ __launch_bounds__(256)
void proj_ce_partial(const float* __restrict__ x, const float* __restrict__ W,
                     const float* __restrict__ b, const int* __restrict__ targ,
                     float* __restrict__ ws_m, float* __restrict__ ws_s,
                     float* __restrict__ ws_tl) {
  __shared__ __align__(16) bf16_t As[2][BLOCK_ROWS * LDS_STRIDE];  // 2 x 10KB
  __shared__ __align__(16) bf16_t Bs[2][SUBW * LDS_STRIDE];        // 2 x 10KB
  __shared__ int   targ_s[BLOCK_ROWS];
  __shared__ float red_m[BLOCK_ROWS][2];
  __shared__ float red_s[BLOCK_ROWS][2];

  const float NEG_INF = -__builtin_inff();

  const int tid  = threadIdx.x;
  const int wave = tid >> 5;
  const int lane = tid & 31;
  const int mw   = wave >> 1;        // 0..3 : which 32-row group
  const int nw   = wave & 1;         // 0..1 : which 64-col group
  const int h    = lane >> 4;        // lane half
  const int ln   = lane & 15;

  const int chunk = blockIdx.x;      // vocab chunk (CHUNK cols)
  const int row0  = blockIdx.y * BLOCK_ROWS;

  if (tid < BLOCK_ROWS) targ_s[tid] = targ[row0 + tid];
  __syncthreads();

  // --- fill-task decomposition: each thread owns one half-row of A and B ---
  const int fr = tid >> 1;            // slab row (A) / slab col (B): 0..127
  const int kh = (tid & 1) * 16;      // which 16-K half of the 32-K slab
  const float* srcA = x + (size_t)(row0 + fr) * K_DIM + kh;
  bf16_t* dstA[2] = { &As[0][fr * LDS_STRIDE + kh], &As[1][fr * LDS_STRIDE + kh] };
  bf16_t* dstB[2] = { &Bs[0][fr * LDS_STRIDE + kh], &Bs[1][fr * LDS_STRIDE + kh] };

  // --- LDS read offsets (bf16 elements) ---
  // A tile mt: lane row = mw*32 + mt*16 + ln; elems[0..7]=K 8h.., [8..15]=K 16+8h..
  const int arow0 = (mw * 32 +  0 + ln) * LDS_STRIDE;
  const int arow1 = (mw * 32 + 16 + ln) * LDS_STRIDE;
  // B tile nt: lane col = nw*64 + nt*16 + ln; elems[0..7]=K 8h.., [8..15]=K 16+8h..
  const int bcol0 = (nw * 64 + ln) * LDS_STRIDE + 8 * h;

  int tloc[2][8];
#pragma unroll
  for (int mt = 0; mt < 2; ++mt)
#pragma unroll
    for (int r = 0; r < 8; ++r)
      tloc[mt][r] = targ_s[mw * 32 + mt * 16 + h * 8 + r];

  float mrun[2][8], srun[2][8];
#pragma unroll
  for (int mt = 0; mt < 2; ++mt)
#pragma unroll
    for (int r = 0; r < 8; ++r) { mrun[mt][r] = NEG_INF; srun[mt][r] = 0.f; }

  for (int sub = 0; sub < NSUB; ++sub) {
    const int cbase = chunk * CHUNK + sub * SUBW;
    const int wcol  = cbase + fr;
    const int wcl   = wcol < V_SIZE ? wcol : (V_SIZE - 1);  // clamp for safe loads
    const float* srcB = W + (size_t)wcl * K_DIM + kh;

    int col[4];
#pragma unroll
    for (int nt = 0; nt < 4; ++nt) col[nt] = cbase + nw * 64 + nt * 16 + ln;

    v8f acc[2][4];
#pragma unroll
    for (int mt = 0; mt < 2; ++mt)
#pragma unroll
      for (int nt = 0; nt < 4; ++nt) acc[mt][nt] = vzero8();

    // prologue: fill buffer 0 with K-slab 0
    store_pack(dstA[0], pack16(load16(srcA)));
    store_pack(dstB[0], pack16(load16(srcB)));
    __syncthreads();

    int buf = 0;
#pragma unroll 2
    for (int k0 = 0; k0 < K_DIM; k0 += 32) {
      const bool more = (k0 + 32) < K_DIM;
      F16R ra, rb;
      if (more) {   // issue next slab's global loads before the math
        ra = load16(srcA + k0 + 32);
        rb = load16(srcB + k0 + 32);
        __builtin_prefetch(srcB + k0 + 96, 0, 1);   // global_prefetch_b8
        __builtin_prefetch(srcA + k0 + 96, 0, 1);
      }

      const bf16_t* Ab = As[buf];
      const bf16_t* Bb = Bs[buf];
      v16bf a0 = lds_read(Ab + arow0 + 8 * h, Ab + arow0 + 16 + 8 * h);
      v16bf a1 = lds_read(Ab + arow1 + 8 * h, Ab + arow1 + 16 + 8 * h);
#pragma unroll
      for (int nt = 0; nt < 4; ++nt) {
        const bf16_t* bp = Bb + bcol0 + nt * 16 * LDS_STRIDE;
        v16bf bm = lds_read(bp, bp + 16);
        acc[0][nt] = wmma_bf16(a0, bm, acc[0][nt]);
        acc[1][nt] = wmma_bf16(a1, bm, acc[1][nt]);
      }

      if (more) {   // convert once, store to the other buffer
        store_pack(buf ? dstA[0] : dstA[1], pack16(ra));
        store_pack(buf ? dstB[0] : dstB[1], pack16(rb));
      }
      __syncthreads();
      buf ^= 1;
    }

    // ---- bias, vocab-edge mask, target capture, per-row max ----
    float msub[2][8];
#pragma unroll
    for (int mt = 0; mt < 2; ++mt)
#pragma unroll
      for (int r = 0; r < 8; ++r) msub[mt][r] = NEG_INF;

#pragma unroll
    for (int nt = 0; nt < 4; ++nt) {
      const bool  cv   = col[nt] < V_SIZE;
      const float bias = cv ? b[col[nt]] : 0.f;
#pragma unroll
      for (int mt = 0; mt < 2; ++mt)
#pragma unroll
        for (int r = 0; r < 8; ++r) {
          float v = cv ? acc[mt][nt][r] + bias : NEG_INF;
          acc[mt][nt][r] = v;
          if (cv && col[nt] == tloc[mt][r])   // unique lane grid-wide per row
            ws_tl[row0 + mw * 32 + mt * 16 + h * 8 + r] = v;
          msub[mt][r] = fmaxf(msub[mt][r], v);
        }
    }
    // reduce max across the 16 lanes of each half (rows live per half)
#pragma unroll
    for (int mt = 0; mt < 2; ++mt)
#pragma unroll
      for (int r = 0; r < 8; ++r)
#pragma unroll
        for (int off = 8; off > 0; off >>= 1)
          msub[mt][r] = fmaxf(msub[mt][r], __shfl_xor(msub[mt][r], off, 32));

    // ---- per-row sum-exp ----
    float ssub[2][8];
#pragma unroll
    for (int mt = 0; mt < 2; ++mt)
#pragma unroll
      for (int r = 0; r < 8; ++r) ssub[mt][r] = 0.f;
#pragma unroll
    for (int nt = 0; nt < 4; ++nt)
#pragma unroll
      for (int mt = 0; mt < 2; ++mt)
#pragma unroll
        for (int r = 0; r < 8; ++r) {
          float v = acc[mt][nt][r];
          if (v > NEG_INF) ssub[mt][r] += __expf(v - msub[mt][r]);
        }
#pragma unroll
    for (int mt = 0; mt < 2; ++mt)
#pragma unroll
      for (int r = 0; r < 8; ++r)
#pragma unroll
        for (int off = 8; off > 0; off >>= 1)
          ssub[mt][r] += __shfl_xor(ssub[mt][r], off, 32);

    // ---- online merge into running (m, s) ----
#pragma unroll
    for (int mt = 0; mt < 2; ++mt)
#pragma unroll
      for (int r = 0; r < 8; ++r) {
        if (msub[mt][r] > NEG_INF) {
          float nm = fmaxf(mrun[mt][r], msub[mt][r]);
          srun[mt][r] = srun[mt][r] * __expf(mrun[mt][r] - nm)
                      + ssub[mt][r] * __expf(msub[mt][r] - nm);
          mrun[mt][r] = nm;
        }
      }
  }

  // ---- combine the two N-waves per row via LDS, write chunk partials ----
  if (ln == 0) {
#pragma unroll
    for (int mt = 0; mt < 2; ++mt)
#pragma unroll
      for (int r = 0; r < 8; ++r) {
        int lr = mw * 32 + mt * 16 + h * 8 + r;
        red_m[lr][nw] = mrun[mt][r];
        red_s[lr][nw] = srun[mt][r];
      }
  }
  __syncthreads();
  if (tid < BLOCK_ROWS) {
    float m0 = red_m[tid][0], m1 = red_m[tid][1];
    float s0 = red_s[tid][0], s1 = red_s[tid][1];
    float M = fmaxf(m0, m1);
    float S = 0.f;
    if (m0 > NEG_INF) S += s0 * __expf(m0 - M);
    if (m1 > NEG_INF) S += s1 * __expf(m1 - M);
    size_t idx = (size_t)(row0 + tid) * NCHUNKS + chunk;
    ws_m[idx] = M;
    ws_s[idx] = S;
  }
}

// ---------------------------------------------------------------------------
// Kernel 2: merge chunk partials per row -> nll; deterministic block sums.
// ---------------------------------------------------------------------------
__global__ __launch_bounds__(256)
void ce_reduce(const float* __restrict__ ws_m, const float* __restrict__ ws_s,
               const float* __restrict__ ws_tl, const int* __restrict__ targ,
               float* __restrict__ ws_blk) {
  const float NEG_INF = -__builtin_inff();
  const int tid = threadIdx.x;
  const int row = blockIdx.x * 256 + tid;

  float nll = 0.f, valid = 0.f;
  {
    const float* mrow = ws_m + (size_t)row * NCHUNKS;
    const float* srow = ws_s + (size_t)row * NCHUNKS;
    float M = NEG_INF;
    for (int c = 0; c < NCHUNKS; ++c) M = fmaxf(M, mrow[c]);
    float S = 0.f;
    for (int c = 0; c < NCHUNKS; ++c) {
      float m = mrow[c];
      if (m > NEG_INF) S += srow[c] * __expf(m - M);
    }
    int t = targ[row];
    if (t != IGNORE_IDX) {
      valid = 1.f;
      nll = __logf(S) + M - ws_tl[row];
    }
  }

  __shared__ float sm[256], sv[256];
  sm[tid] = nll; sv[tid] = valid;
  __syncthreads();
  for (int s = 128; s > 0; s >>= 1) {
    if (tid < s) { sm[tid] += sm[tid + s]; sv[tid] += sv[tid + s]; }
    __syncthreads();
  }
  if (tid == 0) {
    ws_blk[blockIdx.x * 2 + 0] = sm[0];
    ws_blk[blockIdx.x * 2 + 1] = sv[0];
  }
}

__global__ void ce_finalize(const float* __restrict__ ws_blk, float* __restrict__ out) {
  float num = 0.f, cnt = 0.f;
  for (int i = 0; i < RED_BLOCKS; ++i) { num += ws_blk[i * 2]; cnt += ws_blk[i * 2 + 1]; }
  out[0] = num / fmaxf(cnt, 1.f);
}

// ---------------------------------------------------------------------------
extern "C" void kernel_launch(void* const* d_in, const int* in_sizes, int n_in,
                              void* d_out, int out_size, void* d_ws, size_t ws_size,
                              hipStream_t stream) {
  const float* x    = (const float*)d_in[0];
  const float* W    = (const float*)d_in[1];
  const float* b    = (const float*)d_in[2];
  const int*   targ = (const int*)d_in[3];
  float*       out  = (float*)d_out;

  // workspace layout (floats): m[N*NCHUNKS] | s[N*NCHUNKS] | tl[N] | blk[32*2]
  float* ws_m  = (float*)d_ws;
  float* ws_s  = ws_m + (size_t)N_ROWS * NCHUNKS;
  float* ws_tl = ws_s + (size_t)N_ROWS * NCHUNKS;
  float* ws_blk = ws_tl + N_ROWS;

  dim3 grid(NCHUNKS, NROWBLKS);
  proj_ce_partial<<<grid, 256, 0, stream>>>(x, W, b, targ, ws_m, ws_s, ws_tl);
  ce_reduce<<<RED_BLOCKS, 256, 0, stream>>>(ws_m, ws_s, ws_tl, targ, ws_blk);
  ce_finalize<<<1, 1, 0, stream>>>(ws_blk, out);
}
